// SelfAttentionHead_75076028334813
// MI455X (gfx1250) — matrix operand order repaired and checked by reading, not compiled, not run on hardware
//
#include <hip/hip_runtime.h>
#include <hip/hip_fp16.h>

typedef __attribute__((ext_vector_type(16))) _Float16 v16h;
typedef __attribute__((ext_vector_type(8)))  _Float16 v8h;
typedef __attribute__((ext_vector_type(8)))  float    v8f;
typedef __attribute__((ext_vector_type(4)))  unsigned int v4u;
typedef __attribute__((ext_vector_type(4)))  int      v4i;
typedef __attribute__((ext_vector_type(8)))  int      v8i;

#define EDIM 1024
#define HDIM 64
#define TSEQ 4096
#define NBATCH 4
#define KSTEPS (EDIM / 32)
#define LOG2E 1.4426950408889634f

// ---------------------------------------------------------------------------
// WMMA helpers (CDNA5 wave32, V_WMMA_F32_16X16X32_F16)
// ---------------------------------------------------------------------------
static __device__ __forceinline__ v8f wmma_f16(v16h a, v16h b, v8f c) {
  return __builtin_amdgcn_wmma_f32_16x16x32_f16(
      /*neg_a=*/false, a, /*neg_b=*/false, b,
      /*c_mod=*/(short)0, c, /*reuse_a=*/false, /*reuse_b=*/false);
}

// A-matrix 16x32 f16 fragment from row-major [M][K] memory.
static __device__ __forceinline__ v16h load_a_f16(const _Float16* base, int ld, int lane) {
  const int n = lane & 15, hi = lane >> 4;
  const _Float16* p = base + n * ld + hi * 8;
  v8h lo = *(const v8h*)p;          // b128
  v8h hv = *(const v8h*)(p + 16);   // b128
  v16h f;
#pragma unroll
  for (int i = 0; i < 8; ++i) { f[i] = lo[i]; f[i + 8] = hv[i]; }
  return f;
}

// B-matrix 32x16 f16 fragment from row-major B^T = [N][K] memory.
static __device__ __forceinline__ v16h load_b_f16(const _Float16* baseT, int ld, int lane) {
  const int n = lane & 15, hi = lane >> 4;
  const _Float16* p = baseT + n * ld + hi * 8;
  v8h lo = *(const v8h*)p;          // b128
  v8h hv = *(const v8h*)(p + 16);   // b128
  v16h f;
#pragma unroll
  for (int i = 0; i < 8; ++i) { f[i] = lo[i]; f[i + 8] = hv[i]; }
  return f;
}

// B-fragment from the LDS-staged weight slice (192 rows x 32 halves).
static __device__ __forceinline__ v16h load_b_lds(const _Float16* slice,
                                                  int tile, int lane) {
  const int n = lane & 15, hi = lane >> 4;
  const _Float16* p = slice + (tile * 16 + n) * 32 + hi * 8;
  v8h lo = *(const v8h*)p;         // ds_load_b128
  v8h hv = *(const v8h*)(p + 16);  // ds_load_b128
  v16h f;
#pragma unroll
  for (int i = 0; i < 8; ++i) { f[i] = lo[i]; f[i + 8] = hv[i]; }
  return f;
}

// A-fragment built from fp32 source (x), converted to f16 in-register.
static __device__ __forceinline__ v16h load_a_f32cvt(const float* base, int ld, int lane) {
  const int n = lane & 15, hi = lane >> 4;
  const float* p = base + n * ld + hi * 8;
  v16h f;
#pragma unroll
  for (int i = 0; i < 8; ++i) {
    f[i]     = (_Float16)p[i];        // K = hi*8 + i
    f[i + 8] = (_Float16)p[16 + i];   // K = 16 + hi*8 + i
  }
  return f;
}

// ---------------------------------------------------------------------------
// Tensor Data Mover: DMA a [192 rows x 32 cols] f16 tile (the K-slice of all
// three weight matrices, row stride EDIM) from global into LDS, packed as
// 192 consecutive 32-half rows. D# built per CDNA5 ISA ch.8.
// ---------------------------------------------------------------------------
static __device__ __forceinline__ void tdm_load_wslice(const _Float16* gsrc,
                                                       _Float16* ldst) {
  unsigned long long ga = (unsigned long long)(uintptr_t)gsrc;
  unsigned int lds_off  = (unsigned int)(uintptr_t)ldst;  // low 32 = LDS byte addr

  v4u g0;
  g0.x = 1u;                                          // count=1 (valid user D#)
  g0.y = lds_off;                                     // lds_addr
  g0.z = (unsigned int)(ga & 0xFFFFFFFFu);            // global_addr[31:0]
  g0.w = (unsigned int)((ga >> 32) & 0x01FFFFFFu)     // global_addr[56:32]
         | 0x80000000u;                               // type=2 ("image")

  v8i g1;
  g1[0] = 0x00010000;          // workgroup_mask=0, data_size=1 (2 bytes)
  g1[1] = (int)(1024u << 16);  // tensor_dim0 = 1024 (low 16 bits at [31:16])
  g1[2] = (int)(192u  << 16);  // tensor_dim0 hi=0 | tensor_dim1 = 192 (lo16)
  g1[3] = (int)(32u   << 16);  // tensor_dim1 hi=0 | tile_dim0 = 32
  g1[4] = 192;                 // tile_dim1 = 192, tile_dim2 = 0
  g1[5] = 1024;                // tensor_dim0_stride = 1024 (lo 32)
  g1[6] = 0;                   // stride0 hi | tensor_dim1_stride lo (unused, 2D)
  g1[7] = 0;

  v4i z4 = {0, 0, 0, 0};
#if __clang_major__ >= 23
  v8i z8 = {0, 0, 0, 0, 0, 0, 0, 0};
  __builtin_amdgcn_tensor_load_to_lds(g0, g1, z4, z4, z8, 0);
#else
  __builtin_amdgcn_tensor_load_to_lds(g0, g1, z4, z4, 0);
#endif
}

// ---------------------------------------------------------------------------
// Kernel 0: convert Wq/Wk/Wv fp32 -> f16, packed [3][H][E] row-major
// ---------------------------------------------------------------------------
__global__ void __launch_bounds__(256)
cvt_w_kernel(const float* __restrict__ wq, const float* __restrict__ wk,
             const float* __restrict__ wv, _Float16* __restrict__ wf) {
  int idx = blockIdx.x * blockDim.x + threadIdx.x;
  const int plane_sz = HDIM * EDIM;
  if (idx < 3 * plane_sz) {
    int plane = idx / plane_sz;
    int off   = idx - plane * plane_sz;
    const float* src = (plane == 0) ? wq : (plane == 1) ? wk : wv;
    wf[idx] = (_Float16)src[off];
  }
}

// ---------------------------------------------------------------------------
// Kernel 1: fused QKV projection.
// Weight K-slices (192x32 f16 = 12 KB) are TDM-DMA'd into double-buffered LDS
// once per block (shared by all 8 waves). The 12 B-fragments per K-step are
// software-pipelined out of LDS (load t+1 before WMMA t) so DS latency hides
// behind the previous WMMA.
// ---------------------------------------------------------------------------
__global__ void __launch_bounds__(256)
qkv_kernel(const float* __restrict__ x, const _Float16* __restrict__ wf,
           _Float16* __restrict__ Qf, _Float16* __restrict__ Kf,
           _Float16* __restrict__ Vt) {
  __shared__ _Float16 wlds[2][192 * 32];   // 2 x 12 KB weight slices

  const int lane = threadIdx.x & 31;
  const int wave = threadIdx.x >> 5;
  const int row0 = (blockIdx.x * 8 + wave) * 16;   // global row in [0, 16384)
  const int n = lane & 15, hi = lane >> 4;

  v8f acc[12];
#pragma unroll
  for (int t = 0; t < 12; ++t) {
    v8f z = {0.f, 0.f, 0.f, 0.f, 0.f, 0.f, 0.f, 0.f};
    acc[t] = z;
  }

  // Prime the pipeline: slice 0 into buffer 0.
  if (wave == 0) tdm_load_wslice(wf, &wlds[0][0]);

  for (int i = 0; i < KSTEPS; ++i) {
    const int k0  = i * 32;
    const int cur = i & 1;

    __syncthreads();   // all waves done reading buffer cur^1 from prev iter
    if (wave == 0) {
      if (i + 1 < KSTEPS) {
        tdm_load_wslice(wf + (i + 1) * 32, &wlds[cur ^ 1][0]);
        __builtin_amdgcn_s_wait_tensorcnt(1);  // in-order TDM: slice 'cur' done
      } else {
        __builtin_amdgcn_s_wait_tensorcnt(0);
      }
    }
    __syncthreads();   // buffer 'cur' valid for all waves

    v16h a = load_a_f32cvt(x + (size_t)row0 * EDIM + k0, EDIM, lane);

    const _Float16* slice = &wlds[cur][0];
    v16h bcur = load_b_lds(slice, 0, lane);
#pragma unroll
    for (int t = 0; t < 12; ++t) {
      v16h bnext = (t < 11) ? load_b_lds(slice, t + 1, lane) : bcur;
      acc[t] = wmma_f16(a, bcur, acc[t]);
      bcur = bnext;
    }
  }

  // Store results. C layout: lanes 0-15 -> rows r, lanes 16-31 -> rows r+8.
  const int batch = row0 / TSEQ;
#pragma unroll
  for (int nt = 0; nt < 4; ++nt) {
    const int col = nt * 16 + n;
#pragma unroll
    for (int r = 0; r < 8; ++r) {
      const int row = row0 + r + hi * 8;
      const int t   = row - batch * TSEQ;
      Qf[(size_t)row * HDIM + col] = (_Float16)acc[0 + nt][r];
      Kf[(size_t)row * HDIM + col] = (_Float16)acc[4 + nt][r];
      Vt[(size_t)batch * HDIM * TSEQ + (size_t)col * TSEQ + t] =
          (_Float16)acc[8 + nt][r];
    }
  }
}

// ---------------------------------------------------------------------------
// Kernel 2: causal flash attention, one 16-query tile per wave.
// ---------------------------------------------------------------------------
__global__ void __launch_bounds__(256)
attn_kernel(const _Float16* __restrict__ Qf, const _Float16* __restrict__ Kf,
            const _Float16* __restrict__ Vt, float* __restrict__ out) {
  __shared__ _Float16 plds[8][16][32];   // per-wave 16x32 P staging tile

  const int lane = threadIdx.x & 31;
  const int wave = threadIdx.x >> 5;
  const int n = lane & 15, hi = lane >> 4;

  const int tile  = blockIdx.x + gridDim.x * wave;   // 0..1023
  const int batch = tile >> 8;                       // 256 query tiles / batch
  const int q0    = (tile & 255) * 16;

  const _Float16* Qb = Qf + (size_t)(batch * TSEQ + q0) * HDIM;
  const _Float16* Kb = Kf + (size_t)batch * TSEQ * HDIM;
  const _Float16* Vb = Vt + (size_t)batch * HDIM * TSEQ;

  v16h qa0 = load_a_f16(Qb + 0,  HDIM, lane);
  v16h qa1 = load_a_f16(Qb + 32, HDIM, lane);

  v8f o[4];
  float mrow[8], lrow[8];
#pragma unroll
  for (int nt = 0; nt < 4; ++nt) {
    v8f z = {0.f, 0.f, 0.f, 0.f, 0.f, 0.f, 0.f, 0.f};
    o[nt] = z;
  }
#pragma unroll
  for (int r = 0; r < 8; ++r) { mrow[r] = -1e30f; lrow[r] = 0.f; }

  const float scale = 0.03125f;  // 1/sqrt(1024)
  const int qmax = q0 + 15;

  for (int kbase = 0; kbase <= qmax; kbase += 32) {
    // Prefetch next key block (K rows + V^T segments) into cache hierarchy.
    if (kbase + 32 <= qmax) {
      __builtin_prefetch(Kb + (size_t)(kbase + 32 + lane) * HDIM, 0, 3);
      __builtin_prefetch(Vb + (size_t)lane * TSEQ + kbase + 32, 0, 3);
      __builtin_prefetch(Vb + (size_t)(32 + lane) * TSEQ + kbase + 32, 0, 3);
    }

    // Issue all four K B-fragment loads for this 32-key block up front.
    const _Float16* kt0 = Kb + (size_t)kbase * HDIM;
    const _Float16* kt1 = Kb + (size_t)(kbase + 16) * HDIM;
    v16h kb00 = load_b_f16(kt0 + 0,  HDIM, lane);
    v16h kb01 = load_b_f16(kt0 + 32, HDIM, lane);
    v16h kb10 = load_b_f16(kt1 + 0,  HDIM, lane);
    v16h kb11 = load_b_f16(kt1 + 32, HDIM, lane);

    v8f z0 = {0.f, 0.f, 0.f, 0.f, 0.f, 0.f, 0.f, 0.f};
    v8f sc0 = wmma_f16(qa0, kb00, z0);
    sc0 = wmma_f16(qa1, kb01, sc0);
    v8f sc1 = wmma_f16(qa0, kb10, z0);
    sc1 = wmma_f16(qa1, kb11, sc1);

    // Issue the V^T B-fragment loads now; their latency hides behind softmax.
    v16h vb[4];
#pragma unroll
    for (int nt = 0; nt < 4; ++nt)
      vb[nt] = load_b_f16(Vb + (size_t)(nt * 16) * TSEQ + kbase, TSEQ, lane);

    // Scale + causal mask.
    float s[2][8];
    const int key0 = kbase + n;
#pragma unroll
    for (int r = 0; r < 8; ++r) {
      const int q = q0 + r + hi * 8;
      s[0][r] = (key0      <= q) ? sc0[r] * scale : -1e30f;
      s[1][r] = (key0 + 16 <= q) ? sc1[r] * scale : -1e30f;
    }

    // Row max across the 16 columns held by each 16-lane half-group.
    float mx[8];
#pragma unroll
    for (int r = 0; r < 8; ++r) mx[r] = fmaxf(s[0][r], s[1][r]);
#pragma unroll
    for (int xm = 1; xm < 16; xm <<= 1)
#pragma unroll
      for (int r = 0; r < 8; ++r) mx[r] = fmaxf(mx[r], __shfl_xor(mx[r], xm, 32));

    float alpha[8];
#pragma unroll
    for (int r = 0; r < 8; ++r) {
      float mn = fmaxf(mrow[r], mx[r]);
      alpha[r] = __builtin_amdgcn_exp2f((mrow[r] - mn) * LOG2E);
      mrow[r] = mn;
    }

    // P = exp(S - m), f16-stage into LDS, accumulate row sums.
    float rs[8];
#pragma unroll
    for (int r = 0; r < 8; ++r) {
      const int row = r + hi * 8;
      float p0 = __builtin_amdgcn_exp2f((s[0][r] - mrow[r]) * LOG2E);
      float p1 = __builtin_amdgcn_exp2f((s[1][r] - mrow[r]) * LOG2E);
      rs[r] = p0 + p1;
      plds[wave][row][n]      = (_Float16)p0;
      plds[wave][row][16 + n] = (_Float16)p1;
    }
#pragma unroll
    for (int xm = 1; xm < 16; xm <<= 1)
#pragma unroll
      for (int r = 0; r < 8; ++r) rs[r] += __shfl_xor(rs[r], xm, 32);
#pragma unroll
    for (int r = 0; r < 8; ++r) lrow[r] = lrow[r] * alpha[r] + rs[r];

    // Rescale O by alpha.
#pragma unroll
    for (int nt = 0; nt < 4; ++nt)
#pragma unroll
      for (int r = 0; r < 8; ++r) o[nt][r] *= alpha[r];

    // Reload P as an A-fragment (wave-private LDS; DS ops are in-order).
    const _Float16* pp = &plds[wave][n][hi * 8];
    v8h plo = *(const v8h*)pp;
    v8h phi = *(const v8h*)(pp + 16);
    v16h pa;
#pragma unroll
    for (int i = 0; i < 8; ++i) { pa[i] = plo[i]; pa[i + 8] = phi[i]; }

    // O += P(16x32) @ V(32x64); V fragments already in flight.
#pragma unroll
    for (int nt = 0; nt < 4; ++nt) o[nt] = wmma_f16(pa, vb[nt], o[nt]);
  }

  // Epilogue: out[b][q0+row][col] = O / l  (fp32).
  float* ob = out + (size_t)(batch * TSEQ + q0) * HDIM;
  float inv[8];
#pragma unroll
  for (int r = 0; r < 8; ++r) inv[r] = 1.0f / lrow[r];
#pragma unroll
  for (int nt = 0; nt < 4; ++nt) {
    const int col = nt * 16 + n;
#pragma unroll
    for (int r = 0; r < 8; ++r) {
      const int row = r + hi * 8;
      ob[(size_t)row * HDIM + col] = o[nt][r] * inv[r];
    }
  }
}

// ---------------------------------------------------------------------------
// Host launch
// ---------------------------------------------------------------------------
extern "C" void kernel_launch(void* const* d_in, const int* in_sizes, int n_in,
                              void* d_out, int out_size, void* d_ws, size_t ws_size,
                              hipStream_t stream) {
  const float* x  = (const float*)d_in[0];
  const float* wq = (const float*)d_in[1];
  const float* wk = (const float*)d_in[2];
  const float* wv = (const float*)d_in[3];
  float* out = (float*)d_out;

  char* ws = (char*)d_ws;
  // ws layout (bytes):
  //   [0, 393216)          Wqkv f16   [3][64][1024]
  //   [393216, 2490368)    Q f16      [16384][64]
  //   [2490368, 4587520)   K f16      [16384][64]
  //   [4587520, 6684672)   V^T f16    [4][64][4096]
  _Float16* wf = (_Float16*)(ws);
  _Float16* Qf = (_Float16*)(ws + 393216);
  _Float16* Kf = (_Float16*)(ws + 2490368);
  _Float16* Vt = (_Float16*)(ws + 4587520);

  cvt_w_kernel<<<dim3(768), dim3(256), 0, stream>>>(wq, wk, wv, wf);
  qkv_kernel<<<dim3(128), dim3(256), 0, stream>>>(x, wf, Qf, Kf, Vt);
  attn_kernel<<<dim3(128), dim3(256), 0, stream>>>(Qf, Kf, Vt, out);
}
